// AMPBlock0_71262097375779
// MI455X (gfx1250) — matrix operand order repaired and checked by reading, not compiled, not run on hardware
//
#include <hip/hip_runtime.h>

// Problem constants (match reference setup_inputs)
#define BB 8
#define CC 512
#define TT 8192
#define CONVK 3
#define FIRK 12

// Interleaved activation layout: [row = b*16 + c/32][t-slot = t+1][ci = c%32] bf16
// with zeroed slots at t=-1 and t=T (conv zero padding comes for free).
#define NROW (BB * 16)
#define RSLOTS (TT + 2)

// Conv LDS tiling
#define PADW 48     // padded LDS row stride in elements (96B: 16B-aligned, spreads banks)
#define XT 258      // x tile t-slots: [tb-1, tb+256]

typedef __attribute__((ext_vector_type(16))) __bf16 v16bf;
typedef __attribute__((ext_vector_type(8)))  float  v8f;

union Frag { v16bf bf; unsigned int u[8]; };

static __device__ __forceinline__ unsigned short f2bf(float f) {
  unsigned int u = __builtin_bit_cast(unsigned int, f);
  u += 0x7FFFu + ((u >> 16) & 1u);   // round-to-nearest-even
  return (unsigned short)(u >> 16);
}
static __device__ __forceinline__ float bf2f(unsigned short h) {
  return __builtin_bit_cast(float, ((unsigned int)h) << 16);
}
static __device__ __forceinline__ size_t iidx(int row, int t, int ci) {
  return ((size_t)row * RSLOTS + (t + 1)) * 32 + ci;
}

// ---------------------------------------------------------------------------
// Tensor Data Mover (6-arg builtin on this toolchain). D# per ISA 08 §8.3-8.6.
// Pads LDS dest 32B after every 64B of payload (row stride 64B -> 96B).
// ---------------------------------------------------------------------------
#if __has_builtin(__builtin_amdgcn_tensor_load_to_lds)
#define HAVE_TDM 1
typedef __attribute__((ext_vector_type(4))) unsigned int tdm_v4u;
typedef __attribute__((ext_vector_type(8))) int tdm_v8i;
typedef __attribute__((ext_vector_type(4))) int tdm_v4i;

static __device__ __forceinline__ void tdm_load(
    unsigned lds_byte_addr, const void* gptr,
    unsigned ds_code,                       // 0:1B 1:2B 2:4B 3:8B elements
    unsigned tile0, unsigned tile1, unsigned tile2,   // tile dims (== tensor dims)
    unsigned long long s0, unsigned long long s1)     // dim0/dim1 strides (elem units)
{
  unsigned long long ga = (unsigned long long)(size_t)gptr;
  tdm_v4u g0;
  g0.x = 1u;                                              // count=1, user mode
  g0.y = lds_byte_addr;                                   // LDS dest (bytes)
  g0.z = (unsigned)ga;                                    // global_addr[31:0]
  g0.w = (unsigned)((ga >> 32) & 0x01FFFFFFull) | (2u << 30);  // addr[56:32] | type=2
  tdm_v8i g1;
  // data_size | pad_enable | pad_interval=16dw (64B) | pad_amount=8dw (32B)
  g1[0] = (int)((ds_code << 16) | (1u << 20) | (3u << 22) | (7u << 25));
  g1[1] = (int)((tile0 & 0xFFFFu) << 16);                          // tensor_dim0 lo
  g1[2] = (int)((tile0 >> 16) | ((tile1 & 0xFFFFu) << 16));        // td0 hi | td1 lo
  g1[3] = (int)((tile1 >> 16) | ((tile0 & 0xFFFFu) << 16));        // td1 hi | tile_dim0
  g1[4] = (int)((tile1 & 0xFFFFu) | ((tile2 & 0xFFFFu) << 16));    // tile_dim1|tile_dim2
  g1[5] = (int)(s0 & 0xFFFFFFFFull);                               // dim0 stride lo
  g1[6] = (int)(((s0 >> 32) & 0xFFFFull) | ((s1 & 0xFFFFull) << 16));
  g1[7] = (int)((s1 >> 16) & 0xFFFFFFFFull);
  tdm_v4i g2;
  g2[0] = (int)(tile2 ? tile2 : 1);   // tensor_dim2
  g2[1] = 0; g2[2] = 0; g2[3] = 0;
  tdm_v4i g3; g3[0] = 0; g3[1] = 0; g3[2] = 0; g3[3] = 0;
  tdm_v8i g4;
  g4[0] = 0; g4[1] = 0; g4[2] = 0; g4[3] = 0;
  g4[4] = 0; g4[5] = 0; g4[6] = 0; g4[7] = 0;
  __builtin_amdgcn_tensor_load_to_lds(g0, g1, g2, g3, g4, 0);
}
#endif

static __device__ __forceinline__ void tensor_fence() {
#ifdef HAVE_TDM
#if __has_builtin(__builtin_amdgcn_s_wait_tensorcnt)
  __builtin_amdgcn_s_wait_tensorcnt(0);
#else
  asm volatile("s_wait_tensorcnt 0x0" ::: "memory");
#endif
#endif
}

// ---------------------------------------------------------------------------
// Weight transform: w[co][ci][k] f32 -> wt[k][co][ci] bf16
// ---------------------------------------------------------------------------
__global__ __launch_bounds__(256) void wcvt_kernel(const float* __restrict__ w,
                                                   unsigned short* __restrict__ wt) {
  int i = blockIdx.x * 256 + threadIdx.x;
  if (i >= CONVK * CC * CC) return;
  int k  = i / (CC * CC);
  int r  = i - k * CC * CC;
  int co = r / CC;
  int ci = r - co * CC;
  wt[i] = f2bf(w[(co * CC + ci) * CONVK + k]);
}

// ---------------------------------------------------------------------------
// x transpose/convert: x[b][c][t] f32 -> interleaved bf16 (+ zero pad slots)
// ---------------------------------------------------------------------------
__global__ __launch_bounds__(256) void xcvt_kernel(const float* __restrict__ x,
                                                   unsigned short* __restrict__ out) {
  __shared__ unsigned short lt[256 * 33];
  const int tid = threadIdx.x;
  const int t0  = blockIdx.x * 256;
  const int blk = blockIdx.y;
  const int b   = blockIdx.z;
  const int row = b * 16 + blk;
#pragma unroll 4
  for (int i = 0; i < 32; ++i) {
    float v = x[((size_t)b * CC + blk * 32 + i) * TT + t0 + tid];
    lt[tid * 33 + i] = f2bf(v);
  }
  __syncthreads();
#pragma unroll
  for (int j = 0; j < 16; ++j) {
    int e2  = tid + j * 256;          // u32 chunks over 256t x 32ci
    int tt  = e2 >> 4;
    int cip = (e2 & 15) * 2;
    unsigned int v = (unsigned int)lt[tt * 33 + cip] |
                     ((unsigned int)lt[tt * 33 + cip + 1] << 16);
    *(unsigned int*)&out[iidx(row, t0 + tt, cip)] = v;
  }
  if (t0 == 0 && tid < 32) out[iidx(row, -1, tid)] = 0;
  if (t0 + 256 == TT && tid < 32) out[iidx(row, TT, tid)] = 0;
}

// ---------------------------------------------------------------------------
// Fused Activation1d on interleaved bf16 (32 channels per block, 128 t chunk):
//   up[2t]   = 2*sum_a filt[11-2a]*x[clamp(t-3+a)]
//   up[2t+1] = 2*sum_a filt[10-2a]*x[clamp(t-2+a)]
//   s = up + (1/(e^lb+1e-9)) * sin(up*e^la)^2
//   out[t]   = sum_j filt[j]*s[clamp(2t-5+j)]
// ---------------------------------------------------------------------------
#define ATC 128
#define AXN (ATC + 13)
#define ASN (2 * ATC + 12)

__global__ __launch_bounds__(256) void act_kernel(
    const unsigned short* __restrict__ in,    // interleaved bf16
    const float* __restrict__ lalpha, const float* __restrict__ lbeta,
    const float* __restrict__ ufilt, const float* __restrict__ dfilt,
    unsigned short* __restrict__ out)         // interleaved bf16
{
  __shared__ float xl[AXN * 32];
  __shared__ float sl[ASN * 32];
  const int tid = threadIdx.x;
  const int t0  = blockIdx.x * ATC;
  const int blk = blockIdx.y;
  const int b   = blockIdx.z;
  const int row = b * 16 + blk;

  float uf[FIRK], df[FIRK];
#pragma unroll
  for (int j = 0; j < FIRK; ++j) { uf[j] = ufilt[j]; df[j] = dfilt[j]; }
  float fe[6], fo[6];
#pragma unroll
  for (int a = 0; a < 6; ++a) { fe[a] = 2.f * uf[11 - 2 * a]; fo[a] = 2.f * uf[10 - 2 * a]; }

  // Stage x window [t0-6, t0+ATC+6] (replicate clamp), u32 granularity.
  for (int e2 = tid; e2 < AXN * 16; e2 += 256) {
    int j   = e2 >> 4;
    int cip = (e2 & 15) * 2;
    int t   = min(max(t0 - 6 + j, 0), TT - 1);
    unsigned int v = *(const unsigned int*)&in[iidx(row, t, cip)];
    xl[j * 32 + cip]     = bf2f((unsigned short)v);
    xl[j * 32 + cip + 1] = bf2f((unsigned short)(v >> 16));
  }
  __syncthreads();

  const int ci = tid & 31;   // constant per thread in all strided loops below
  const float alpha = __expf(lalpha[blk * 32 + ci]);
  const float rbeta = 1.0f / (__expf(lbeta[blk * 32 + ci]) + 1e-9f);
  const int u0 = 2 * t0 - 5;

  for (int e = tid; e < ASN * 32; e += 256) {
    int j  = e >> 5;
    int u  = u0 + j;
    int r  = u & 1;
    int tp = u >> 1;
    int base = tp - t0 + 3;
    float up = 0.f;
    if (r == 0) {
#pragma unroll
      for (int a = 0; a < 6; ++a) up += fe[a] * xl[(base + a) * 32 + ci];
    } else {
#pragma unroll
      for (int a = 0; a < 6; ++a) up += fo[a] * xl[(base + 1 + a) * 32 + ci];
    }
    float sn = __sinf(up * alpha);
    sl[j * 32 + ci] = up + rbeta * sn * sn;
  }
  __syncthreads();

  for (int e = tid; e < ATC * 32; e += 256) {
    int tt = e >> 5;
    float acc = 0.f;
#pragma unroll
    for (int j = 0; j < FIRK; ++j) {
      int u = min(max(2 * (t0 + tt) - 5 + j, 0), 2 * TT - 1);
      acc += df[j] * sl[(u - u0) * 32 + ci];
    }
    out[iidx(row, t0 + tt, ci)] = f2bf(acc);
  }
  if (t0 == 0 && tid < 32) out[iidx(row, -1, tid)] = 0;
  if (t0 + ATC == TT && tid < 32) out[iidx(row, TT, tid)] = 0;
}

// ---------------------------------------------------------------------------
// Implicit-GEMM conv1d (k=3) with bf16 WMMA, f32 accumulate.
// Block 256 thr = 8 waves; block tile 64co x 256t; wave tile 32co x 64t
// (2x4 wmma tiles). K loop: 16 ci-chunks x 3 taps, TDM double-buffered LDS.
// ---------------------------------------------------------------------------
template <bool FINAL>
__global__ __launch_bounds__(256) void conv_kernel(
    const unsigned short* __restrict__ act,   // interleaved bf16
    const unsigned short* __restrict__ wt,    // [k][co][ci] bf16
    const float* __restrict__ bias,           // [C]
    const float* __restrict__ resid,          // x f32 (used iff FINAL)
    void* __restrict__ outp)                  // interleaved bf16 | [B][C][T] f32
{
  __shared__ __align__(16) unsigned short wlds[2][CONVK * 64 * PADW];  // 2x18.4KB
  __shared__ __align__(16) unsigned short xlds[2][XT * PADW];          // 2x24.8KB

  const int tid  = threadIdx.x;
  const int lane = tid & 31;
  const int wid  = tid >> 5;
  const int half = lane >> 4;
  const int lm   = lane & 15;
  const int b    = blockIdx.z;
  const int cob  = blockIdx.y * 64;
  const int tb   = blockIdx.x * 256;
  const int wm   = wid & 1;    // 2 co-groups of 32
  const int wn   = wid >> 1;   // 4 t-groups of 64

  v8f acc[2][4];
#pragma unroll
  for (int mi = 0; mi < 2; ++mi)
#pragma unroll
    for (int ni = 0; ni < 4; ++ni)
#pragma unroll
      for (int r = 0; r < 8; ++r) acc[mi][ni][r] = 0.f;

  auto stage = [&](int cc, int bufi) {
#ifdef HAVE_TDM
    if (tid < 32) {   // wave 0 issues both TDM descriptors
      // weights: 3D tile ci(32) x co(64) x k(3); strides CC, CC*CC elements
      tdm_load((unsigned)(size_t)&wlds[bufi][0], wt + (size_t)cob * CC + cc,
               1u, 32u, 64u, 3u,
               (unsigned long long)CC, (unsigned long long)CC * CC);
      // x: contiguous 1D slab of 258 t-slots x 32 ci (16512B), 8B elements
      tdm_load((unsigned)(size_t)&xlds[bufi][0],
               act + ((size_t)(b * 16 + (cc >> 5)) * RSLOTS + tb) * 32,
               3u, 2064u, 1u, 0u, 2064ull, 0ull);
    }
#else
    for (int q = tid; q < CONVK * 64 * 4; q += 256) {
      int rowid = q >> 2, part = q & 3;
      int k = rowid >> 6, co = rowid & 63;
      *(uint4*)&wlds[bufi][rowid * PADW + part * 8] =
          *(const uint4*)&wt[((size_t)k * CC + cob + co) * CC + cc + part * 8];
    }
    const unsigned short* xs =
        act + ((size_t)(b * 16 + (cc >> 5)) * RSLOTS + tb) * 32;
    for (int q = tid; q < XT * 4; q += 256) {
      int tt = q >> 2, part = q & 3;
      *(uint4*)&xlds[bufi][tt * PADW + part * 8] =
          *(const uint4*)&xs[tt * 32 + part * 8];
    }
#endif
  };

  stage(0, 0);
  tensor_fence();
  __syncthreads();

  int cur = 0;
  for (int cc = 0; cc < CC; cc += 32) {
    if (cc + 32 < CC) stage(cc + 32, cur ^ 1);   // overlap next-chunk DMA

#pragma unroll
    for (int k3 = 0; k3 < CONVK; ++k3) {
      Frag afr[2];
#pragma unroll
      for (int mi = 0; mi < 2; ++mi) {
        int wrow = k3 * 64 + wm * 32 + mi * 16 + lm;
#pragma unroll
        for (int v = 0; v < 8; ++v) {
          int kk = half * 16 + v * 2;
          afr[mi].u[v] = *(const unsigned int*)&wlds[cur][wrow * PADW + kk];
        }
      }
      Frag bfr[4];
#pragma unroll
      for (int ni = 0; ni < 4; ++ni) {
        int tloc = wn * 64 + ni * 16 + lm + k3;
#pragma unroll
        for (int v = 0; v < 8; ++v) {
          int kk = half * 16 + v * 2;
          bfr[ni].u[v] = *(const unsigned int*)&xlds[cur][tloc * PADW + kk];
        }
      }
#pragma unroll
      for (int mi = 0; mi < 2; ++mi)
#pragma unroll
        for (int ni = 0; ni < 4; ++ni)
          acc[mi][ni] = __builtin_amdgcn_wmma_f32_16x16x32_bf16(
              false, afr[mi].bf, false, bfr[ni].bf,
              (short)0, acc[mi][ni], false, false);
    }
    tensor_fence();
    __syncthreads();
    cur ^= 1;
  }

  // Epilogue: VGPR r holds (M = r + 8*half, N = lane&15) of each 16x16 tile.
#pragma unroll
  for (int mi = 0; mi < 2; ++mi) {
    int cog0 = cob + wm * 32 + mi * 16 + half * 8;   // 8 consecutive channels
#pragma unroll
    for (int ni = 0; ni < 4; ++ni) {
      int tg = tb + wn * 64 + ni * 16 + lm;
      if (FINAL) {
#pragma unroll
        for (int r = 0; r < 8; ++r) {
          int cog = cog0 + r;
          size_t off = ((size_t)b * CC + cog) * TT + tg;
          ((float*)outp)[off] = acc[mi][ni][r] + bias[cog] + resid[off];
        }
      } else {
        uint4 pk;
        unsigned int w[4];
#pragma unroll
        for (int p = 0; p < 4; ++p) {
          unsigned short lo = f2bf(acc[mi][ni][2 * p]     + bias[cog0 + 2 * p]);
          unsigned short hi = f2bf(acc[mi][ni][2 * p + 1] + bias[cog0 + 2 * p + 1]);
          w[p] = (unsigned int)lo | ((unsigned int)hi << 16);
        }
        pk.x = w[0]; pk.y = w[1]; pk.z = w[2]; pk.w = w[3];
        size_t off = iidx(b * 16 + (cog0 >> 5), tg, cog0 & 31);
        *(uint4*)&((unsigned short*)outp)[off] = pk;
      }
    }
  }
}

// ---------------------------------------------------------------------------
extern "C" void kernel_launch(void* const* d_in, const int* in_sizes, int n_in,
                              void* d_out, int out_size, void* d_ws, size_t ws_size,
                              hipStream_t stream) {
  const float* x        = (const float*)d_in[0];
  const float* a1_alpha = (const float*)d_in[1];
  const float* a1_beta  = (const float*)d_in[2];
  const float* a2_alpha = (const float*)d_in[3];
  const float* a2_beta  = (const float*)d_in[4];
  const float* c1_w     = (const float*)d_in[5];
  const float* c1_b     = (const float*)d_in[6];
  const float* c2_w     = (const float*)d_in[7];
  const float* c2_b     = (const float*)d_in[8];
  const float* up_f     = (const float*)d_in[9];
  const float* dn_f     = (const float*)d_in[10];

  unsigned short* wt1 = (unsigned short*)d_ws;
  unsigned short* wt2 = wt1 + (size_t)CONVK * CC * CC;
  unsigned short* P   = wt2 + (size_t)CONVK * CC * CC;     // interleaved buffer
  unsigned short* Q   = P + (size_t)NROW * RSLOTS * 32;    // interleaved buffer

  const int nw = CONVK * CC * CC;
  wcvt_kernel<<<(nw + 255) / 256, 256, 0, stream>>>(c1_w, wt1);
  wcvt_kernel<<<(nw + 255) / 256, 256, 0, stream>>>(c2_w, wt2);

  dim3 xgrid(TT / 256, 16, BB);
  dim3 agrid(TT / ATC, 16, BB);
  dim3 cgrid(TT / 256, CC / 64, BB);

  // x f32 -> P (interleaved bf16, pads zeroed)
  xcvt_kernel<<<xgrid, 256, 0, stream>>>(x, P);
  // act1: P -> Q
  act_kernel<<<agrid, 256, 0, stream>>>(P, a1_alpha, a1_beta, up_f, dn_f, Q);
  // conv1: Q -> P (bf16 interleaved; pads in P persist from xcvt)
  conv_kernel<false><<<cgrid, 256, 0, stream>>>(Q, wt1, c1_b, x, (void*)P);
  // act2: P -> Q
  act_kernel<<<agrid, 256, 0, stream>>>(P, a2_alpha, a2_beta, up_f, dn_f, Q);
  // conv2: Q -> d_out (f32, +bias +residual)
  conv_kernel<true><<<cgrid, 256, 0, stream>>>(Q, wt2, c2_b, x, d_out);
}
